// ConvBlock4D_27556510171710
// MI455X (gfx1250) — matrix operand order repaired and checked by reading, not compiled, not run on hardware
//
#include <hip/hip_runtime.h>

// ---------------------------------------------------------------------------
// ConvBlock4D on MI455X (gfx1250, wave32, WMMA).
//
// Roofline: conv1 is 87 GFLOP vs ~100MB traffic -> compute bound; run it on
// v_wmma_f32_16x16x32_f16 (f16 mul, f32 accumulate; ~5e-4 rel err at K=5184).
// x (32MB) << 192MB L2, so per-(dt,dl) re-reads of x are L2 hits.
//
// Kernel 1: repack w1 -> f16 [tap][cout][cin], w2 -> f16 [cout][cin].
// Kernel 2: implicit-GEMM conv1. Block = 256 thr (8 waves) = 128 positions
//           (4 H-rows x 32 W) x 64 couts. For each in-bounds (dt,dl): stage a
//           halo-padded 6x34x64 f16 x-slab [h'][w'][cin] ONCE (zeroed w-halo
//           cols written once at start; zeroed OOB h rows), then for each of
//           the 9 (dh,dw) taps just stage the 8KB weight tile and read A
//           fragments as shifted ds_load_b128 from the slab. All 20 b128 fragment
//           loads are hoisted ahead of the 8 back-to-back WMMAs. Next tap's
//           weight tile is prefetched (global_prefetch_b8) during compute.
// Kernel 3: fused LayerNorm(fp32) + exact GELU + 1x1 conv (8 WMMAs) +
//           bias + residual, scattered store into NCTLHW output.
// ---------------------------------------------------------------------------

typedef __attribute__((ext_vector_type(16))) _Float16 v16h;
typedef __attribute__((ext_vector_type(8)))  float    v8f;
typedef __attribute__((ext_vector_type(4)))  unsigned int u32x4;

union HFrag { v16h v; u32x4 q[2]; };

// strides in x/out (NCTLHW): w=1, h=32, l=1024, t=8192, c=65536, n=4194304
#define XSTRIDE_C 65536
#define XSTRIDE_N 4194304

#define W1P_ELEMS (81 * 64 * 64)   // 331776
#define W2P_ELEMS (64 * 64)        // 4096

// ---------------------------------------------------------------- prep ----
__global__ __launch_bounds__(256) void prep_kernel(
    const float* __restrict__ w1, const float* __restrict__ w2,
    _Float16* __restrict__ w1p, _Float16* __restrict__ w2p)
{
    int i = blockIdx.x * 256 + threadIdx.x;
    if (i < W1P_ELEMS) {
        int ci  = i & 63;
        int co  = (i >> 6) & 63;
        int tap = i >> 12;
        // w1 flat: [co][ci][81 taps]
        w1p[i] = (_Float16)w1[(co * 64 + ci) * 81 + tap];
    } else {
        int j = i - W1P_ELEMS;
        if (j < W2P_ELEMS) w2p[j] = (_Float16)w2[j];
    }
}

// --------------------------------------------------------------- conv1 ----
__global__ __launch_bounds__(256) void conv1_kernel(
    const float* __restrict__ x, const _Float16* __restrict__ w1p,
    const float* __restrict__ b1, float* __restrict__ h1)
{
    // x slab: [h'(6)][w'(34, halo)][cin(64)] f16 = 25.5KB ; weights 8KB
    __shared__ __attribute__((aligned(16))) _Float16 sX[6 * 34 * 64];
    __shared__ __attribute__((aligned(16))) _Float16 sB[64 * 64];

    const int b    = blockIdx.x;          // 1024 blocks = N*T*L*(H/4)
    const int h0   = (b & 7) * 4;
    const int l    = (b >> 3) & 7;
    const int t    = (b >> 6) & 7;
    const int n    = b >> 9;
    const int tid  = threadIdx.x;
    const int lane = tid & 31;
    const int wv   = tid >> 5;

    // zero the w-halo columns (w'=0 and w'=33) once; never rewritten
    for (int e = tid; e < 6 * 2 * 64; e += 256) {
        const int rr   = e >> 7;
        const int side = (e >> 6) & 1;
        const int cin  = e & 63;
        sX[(rr * 34 + side * 33) * 64 + cin] = (_Float16)0.f;
    }

    v8f acc[4] = {};

    const int cin_ld = tid >> 2;          // staging: one cin per 4 threads
    const int q_ld   = tid & 3;           // quarter of a W row
    const int mrow   = wv * 16 + (lane & 15);
    const int half   = lane >> 4;
    const int rpos   = mrow >> 5;         // H-row (0..3) within tile
    const int wpos   = mrow & 31;         // W within tile
    const int col16  = lane & 15;

    for (int dt = 0; dt < 3; ++dt) {
        const int tt = t + dt - 1;
        if ((unsigned)tt >= 8u) continue;
        for (int dl = 0; dl < 3; ++dl) {
            const int ll = l + dl - 1;
            if ((unsigned)ll >= 8u) continue;

            // --- stage halo-padded x slab once per (dt,dl), transposed ---
            {
                const float* xbase =
                    x + ((((n * 64 + cin_ld) * 8 + tt) * 8 + ll) * 1024);
                #pragma unroll
                for (int rr = 0; rr < 6; ++rr) {
                    const int hh   = h0 + rr - 1;
                    const bool hok = (unsigned)hh < 32u;
                    const float* xrow = xbase + hh * 32 + q_ld * 8;
                    _Float16* dst = &sX[(rr * 34 + 1 + q_ld * 8) * 64 + cin_ld];
                    #pragma unroll
                    for (int j = 0; j < 8; ++j) {
                        float v = hok ? xrow[j] : 0.f;
                        dst[j * 64] = (_Float16)v;
                    }
                }
            }

            for (int dh = 0; dh < 3; ++dh) {
                for (int dw = 0; dw < 3; ++dw) {
                    const int tap = ((dt * 3 + dl) * 3 + dh) * 3 + dw;

                    // stage weight tile: 4096 f16 = 512 x 16B
                    {
                        const u32x4* src = (const u32x4*)(w1p + tap * 4096);
                        u32x4* dst = (u32x4*)sB;
                        dst[tid]       = src[tid];
                        dst[tid + 256] = src[tid + 256];
                        // prefetch next tap's tile (global_prefetch_b8)
                        __builtin_prefetch(w1p + (tap + 1) * 4096 + tid * 16, 0, 1);
                    }
                    __syncthreads();

                    // hoist all fragment loads, then 8 back-to-back WMMAs
                    const int abase =
                        ((rpos + dh) * 34 + (wpos + dw)) * 64 + half * 8;
                    HFrag a[2], bf[2][4];
                    #pragma unroll
                    for (int kc = 0; kc < 2; ++kc) {
                        const u32x4* ap = (const u32x4*)&sX[abase + kc * 32];
                        a[kc].q[0] = ap[0];
                        a[kc].q[1] = ap[2];            // +16 f16
                        #pragma unroll
                        for (int nt = 0; nt < 4; ++nt) {
                            const u32x4* bp = (const u32x4*)
                                &sB[(nt * 16 + col16) * 64 + kc * 32 + half * 8];
                            bf[kc][nt].q[0] = bp[0];
                            bf[kc][nt].q[1] = bp[2];
                        }
                    }
                    #pragma unroll
                    for (int kc = 0; kc < 2; ++kc) {
                        #pragma unroll
                        for (int nt = 0; nt < 4; ++nt) {
                            acc[nt] = __builtin_amdgcn_wmma_f32_16x16x32_f16(
                                false, a[kc].v, false, bf[kc][nt].v, (short)0,
                                acc[nt], false, false);
                        }
                    }
                    __syncthreads();
                }
            }
        }
    }

    // --- store h1[pos][c] = acc + b1 (fp32) ---
    const int base_row = n * 65536 + t * 8192 + l * 1024 + h0 * 32;
    const int moff0 = half * 8;
    #pragma unroll
    for (int nt = 0; nt < 4; ++nt) {
        const int c = nt * 16 + col16;
        const float bias = b1[c];
        #pragma unroll
        for (int r = 0; r < 8; ++r) {
            const int m = wv * 16 + moff0 + r;
            h1[(size_t)(base_row + m) * 64 + c] = acc[nt][r] + bias;
        }
    }
}

// ----------------------------------------------- LN + GELU + 1x1 + res ----
__global__ __launch_bounds__(256) void fuse_kernel(
    const float* __restrict__ x, const float* __restrict__ h1,
    const _Float16* __restrict__ w2p, const float* __restrict__ lnw,
    const float* __restrict__ lnb, const float* __restrict__ b2,
    float* __restrict__ out)
{
    __shared__ __attribute__((aligned(16))) _Float16 sG[128 * 64]; // gelu(ln) f16
    __shared__ __attribute__((aligned(16))) _Float16 sB[64 * 64];  // w2 [cout][cin]

    const int b    = blockIdx.x;      // 1024 blocks x 128 positions
    const int tid  = threadIdx.x;
    const int lane = tid & 31;
    const int wv   = tid >> 5;
    const int base_row = b * 128;

    {   // stage w2 tile
        const u32x4* src = (const u32x4*)w2p;
        u32x4* dst = (u32x4*)sB;
        dst[tid]       = src[tid];
        dst[tid + 256] = src[tid + 256];
    }

    {   // LayerNorm + exact GELU: lane pair (2 lanes) per 64-wide row
        const int r  = wv * 16 + (lane >> 1);
        const int hf = lane & 1;
        const float* hrow = h1 + (size_t)(base_row + r) * 64 + hf * 32;
        float vals[32];
        float s = 0.f, ss = 0.f;
        #pragma unroll 8
        for (int j = 0; j < 32; ++j) {
            float v = hrow[j];
            vals[j] = v; s += v; ss += v * v;
        }
        s  += __shfl_xor(s, 1, 32);
        ss += __shfl_xor(ss, 1, 32);
        const float mean = s * (1.f / 64.f);
        const float var  = ss * (1.f / 64.f) - mean * mean;
        const float rstd = rsqrtf(var + 1e-5f);
        _Float16* g = &sG[r * 64 + hf * 32];
        #pragma unroll 8
        for (int j = 0; j < 32; ++j) {
            const int c = hf * 32 + j;
            float hn = (vals[j] - mean) * rstd;
            float y  = hn * lnw[c] + lnb[c];
            float ge = 0.5f * y * (1.f + erff(y * 0.70710678118654752f));
            g[j] = (_Float16)ge;
        }
    }
    __syncthreads();

    // 1x1 conv GEMM: 16 rows x 64 cout per wave, K=64
    v8f acc[4] = {};
    const int mrow = wv * 16 + (lane & 15);
    const int half = lane >> 4;
    const int col16 = lane & 15;

    HFrag a[2], bf[2][4];
    #pragma unroll
    for (int kc = 0; kc < 2; ++kc) {
        const u32x4* ap = (const u32x4*)&sG[mrow * 64 + kc * 32 + half * 8];
        a[kc].q[0] = ap[0];
        a[kc].q[1] = ap[2];
        #pragma unroll
        for (int nt = 0; nt < 4; ++nt) {
            const u32x4* bp =
                (const u32x4*)&sB[(nt * 16 + col16) * 64 + kc * 32 + half * 8];
            bf[kc][nt].q[0] = bp[0];
            bf[kc][nt].q[1] = bp[2];
        }
    }
    #pragma unroll
    for (int kc = 0; kc < 2; ++kc) {
        #pragma unroll
        for (int nt = 0; nt < 4; ++nt) {
            acc[nt] = __builtin_amdgcn_wmma_f32_16x16x32_f16(
                false, a[kc].v, false, bf[kc][nt].v, (short)0, acc[nt],
                false, false);
        }
    }

    // out[n,c,sp] = x[n,c,sp] + b2[c] + acc
    const int moff0 = half * 8;
    #pragma unroll
    for (int nt = 0; nt < 4; ++nt) {
        const int c = nt * 16 + col16;
        const float bias = b2[c];
        #pragma unroll
        for (int r = 0; r < 8; ++r) {
            const int m   = wv * 16 + moff0 + r;
            const int pos = base_row + m;
            const int ni  = pos >> 16;
            const int sp  = pos & 65535;
            const size_t oi = (size_t)ni * XSTRIDE_N + (size_t)c * XSTRIDE_C + sp;
            out[oi] = x[oi] + bias + acc[nt][r];
        }
    }
}

// ------------------------------------------------------------- launch ----
extern "C" void kernel_launch(void* const* d_in, const int* in_sizes, int n_in,
                              void* d_out, int out_size, void* d_ws, size_t ws_size,
                              hipStream_t stream) {
    const float* x   = (const float*)d_in[0];
    const float* w1  = (const float*)d_in[1];
    const float* b1  = (const float*)d_in[2];
    const float* lnw = (const float*)d_in[3];
    const float* lnb = (const float*)d_in[4];
    const float* w2  = (const float*)d_in[5];
    const float* b2  = (const float*)d_in[6];
    float* out = (float*)d_out;

    char* ws = (char*)d_ws;
    float*     h1  = (float*)ws;                              // 32 MiB
    _Float16*  w1p = (_Float16*)(ws + 33554432);              // 663552 B
    _Float16*  w2p = (_Float16*)(ws + 33554432 + 663552);     // 8192 B

    prep_kernel <<<(W1P_ELEMS + W2P_ELEMS) / 256, 256, 0, stream>>>(w1, w2, w1p, w2p);
    conv1_kernel<<<1024, 256, 0, stream>>>(x, w1p, b1, h1);
    fuse_kernel <<<1024, 256, 0, stream>>>(x, h1, w2p, lnw, lnb, b2, out);
}
